// ImplicitModel_11579231830016
// MI455X (gfx1250) — matrix operand order repaired and checked by reading, not compiled, not run on hardware
//
#include <hip/hip_runtime.h>

// ===========================================================================
// Implicit DEQ model for MI455X (gfx1250, wave32, WMMA).
//   out = (C @ X* + D @ U^T)^T,  X* = fixpoint of X = relu(Ap @ X + B@U^T)
//
// Precision: compensated bf16x2 split (v = hi + lo): product = hi*hi + hi*lo
// + lo*hi with f32 WMMA accumulation -> ~1e-5 relative accuracy at 8/3x fewer
// matrix instructions than the fp32 16x16x4 WMMA path. Hot-loop working set
// (~40MB) is L2-resident (192MB), so the loop is matrix-instruction bound.
//
// Data movement: X is kept TRANSPOSED (Xt[n][k], K-contiguous) so every GEMM
// operand is K-contiguous -> all LDS staging is pure 16B chunks, done with
// GLOBAL_LOAD_ASYNC_TO_LDS_B128 (ASYNCcnt) when available, double-buffered in
// LDS with one barrier per 32-deep K tile.
// ===========================================================================

typedef __attribute__((ext_vector_type(16))) __bf16   v16bf_t;
typedef __attribute__((ext_vector_type(8)))  float    v8f_t;
typedef __attribute__((ext_vector_type(4)))  unsigned u32x4_t;

// builtin's parameter type per hipcc diagnostic: vector_size(16) int pointer
typedef int impdeq_i32x4 __attribute__((vector_size(16)));
typedef __attribute__((address_space(1))) impdeq_i32x4 gv4i_t;   // global
typedef __attribute__((address_space(3))) impdeq_i32x4 lv4i_t;   // LDS

union FragBF { v16bf_t v; u32x4_t q[2]; };
union Pack8  { u32x4_t  q; __bf16  h[8]; };

static constexpr int kHidden = 1024;
static constexpr int kInput  = 2048;
static constexpr int kOutput = 1000;
static constexpr int kBatch  = 4096;
static constexpr int kPicardIters = 16;   // contraction ~0.06/iter -> fp32-converged

// ---- CDNA5 async-to-LDS path (probe via __has_builtin, with fallback) -----
#if defined(__has_builtin)
#  if __has_builtin(__builtin_amdgcn_global_load_async_to_lds_b128) && \
      __has_builtin(__builtin_amdgcn_s_wait_asynccnt)
#    define IMPDEQ_ASYNC 1
#  endif
#endif
#ifndef IMPDEQ_ASYNC
#  define IMPDEQ_ASYNC 0
#endif

__device__ __forceinline__ void copy16_to_lds(const __bf16* g, __bf16* l) {
#if IMPDEQ_ASYNC
  __builtin_amdgcn_global_load_async_to_lds_b128((gv4i_t*)g, (lv4i_t*)l, 0, 0);
#else
  *(u32x4_t*)l = *(const u32x4_t*)g;
#endif
}

__device__ __forceinline__ void wait_lds_copies() {
#if IMPDEQ_ASYNC
  __builtin_amdgcn_s_wait_asynccnt(0);
#endif
}

__device__ __forceinline__ void split2bf16(float v, __bf16& h, __bf16& l) {
  h = (__bf16)v;
  l = (__bf16)(v - (float)h);
}

// ---- row-wise L-inf projection of A, fused with bf16x2 split ---------------
__global__ __launch_bounds__(256)
void impdeq_project_split(const float* __restrict__ A,
                          __bf16* __restrict__ hi, __bf16* __restrict__ lo, int n) {
  __shared__ float red[256];
  const int row = blockIdx.x;
  const float* a = A + (size_t)row * n;
  float s = 0.0f;
  for (int k = threadIdx.x; k < n; k += 256) s += fabsf(a[k]);
  red[threadIdx.x] = s;
  __syncthreads();
  for (int off = 128; off > 0; off >>= 1) {
    if (threadIdx.x < off) red[threadIdx.x] += red[threadIdx.x + off];
    __syncthreads();
  }
  const float scale = fminf(1.0f, 0.95f / fmaxf(red[0], 1e-12f));
  for (int k = threadIdx.x; k < n; k += 256) {
    float v = a[k] * scale;
    __bf16 h, l; split2bf16(v, h, l);
    hi[(size_t)row * n + k] = h;
    lo[(size_t)row * n + k] = l;
  }
}

// ---- elementwise f32 -> (bf16 hi, bf16 lo) --------------------------------
__global__ __launch_bounds__(256)
void impdeq_split(const float* __restrict__ src,
                  __bf16* __restrict__ hi, __bf16* __restrict__ lo, size_t cnt) {
  size_t i = (size_t)blockIdx.x * 256 + threadIdx.x;
  const size_t stride = (size_t)gridDim.x * 256;
  for (; i < cnt; i += stride) {
    __bf16 h, l; split2bf16(src[i], h, l);
    hi[i] = h; lo[i] = l;
  }
}

// ---- X0^T = relu(BU)^T, split: XtHi/XtLo[n][k] <- relu(BU[k][n]) ----------
__global__ __launch_bounds__(256)
void impdeq_relu_split_T(const float* __restrict__ src,          // [Kd x Nd]
                         __bf16* __restrict__ hiT, __bf16* __restrict__ loT,
                         int Kd, int Nd) {
  size_t i = (size_t)blockIdx.x * 256 + threadIdx.x;
  const size_t stride = (size_t)gridDim.x * 256;
  const size_t total = (size_t)Kd * Nd;
  for (; i < total; i += stride) {
    const int k = (int)(i & (size_t)(Kd - 1));   // Kd is a power of two (1024)
    const int n = (int)(i >> 10);
    __bf16 h, l; split2bf16(fmaxf(src[(size_t)k * Nd + n], 0.0f), h, l);
    hiT[i] = h;   // i == n*Kd + k
    loT[i] = l;
  }
}

// ---------------------------------------------------------------------------
// Compensated bf16x2 GEMM, all operands K-contiguous ("NT"):
//   out(MxN) = Arow(MxK) * Brow(NxK)^T  [+ bias] [relu]
//   STORE 0 : f32 row-major, ld = N                       (BU, T1)
//   STORE 1 : split bf16 TRANSPOSED, outHi/Lo[col*ldT+row] (Picard X update)
//   STORE 2 : f32 TRANSPOSED, outF[col*ldT+row]            (final output)
// Block 64(M) x 128(N), 8 wave32s as 4(M) x 2(N); wave tile 16 x 64.
// LDS double-buffered; staging via async 16B copies; 1 barrier per K-tile.
// ---------------------------------------------------------------------------
template<bool BIAS, bool RELU, int STORE>
__global__ __launch_bounds__(256)
void impdeq_gemm(const __bf16* __restrict__ Ahi, const __bf16* __restrict__ Alo, int lda,
                 const __bf16* __restrict__ Bhi, const __bf16* __restrict__ Blo, int ldb,
                 const float*  __restrict__ bias,
                 float*  __restrict__ outF,
                 __bf16* __restrict__ outHi, __bf16* __restrict__ outLo,
                 int ldOutT, int M, int N, int K) {
  __shared__ __align__(16) __bf16 sA[2][2][64][40];    // [buf][hi/lo][m][k] 20KB
  __shared__ __align__(16) __bf16 sB[2][2][128][40];   // [buf][hi/lo][n][k] 40KB

  const int tid  = threadIdx.x;
  const int lane = tid & 31;
  const int wave = tid >> 5;
  const int wm0  = (wave & 3) * 16;
  const int wn0  = (wave >> 2) * 64;
  const int half = lane >> 4;
  const int lr   = lane & 15;

  const int m0 = blockIdx.y * 64;
  const int n0 = blockIdx.x * 128;

  // M-edge: pre-zero invalid A rows in BOTH buffers (async copies skip them).
  {
    const int m  = tid >> 2;
    const int kc = (tid & 3) * 8;
    if (m0 + m >= M) {
      const u32x4_t z = {0, 0, 0, 0};
      *(u32x4_t*)&sA[0][0][m][kc] = z;  *(u32x4_t*)&sA[0][1][m][kc] = z;
      *(u32x4_t*)&sA[1][0][m][kc] = z;  *(u32x4_t*)&sA[1][1][m][kc] = z;
    }
  }

  auto stage = [&](int p, int k0) {
    {  // A: 64x32 bf16 per plane, one 16B chunk per thread per plane
      const int m  = tid >> 2;
      const int kc = (tid & 3) * 8;
      const int gr = m0 + m;
      if (gr < M) {
        copy16_to_lds(Ahi + (size_t)gr * lda + k0 + kc, &sA[p][0][m][kc]);
        copy16_to_lds(Alo + (size_t)gr * lda + k0 + kc, &sA[p][1][m][kc]);
      }
    }
    for (int c = tid; c < 512; c += 256) {  // B: 128x32 bf16 per plane
      const int n  = c >> 2;
      const int kc = (c & 3) * 8;
      copy16_to_lds(Bhi + (size_t)(n0 + n) * ldb + k0 + kc, &sB[p][0][n][kc]);
      copy16_to_lds(Blo + (size_t)(n0 + n) * ldb + k0 + kc, &sB[p][1][n][kc]);
    }
  };

  v8f_t acc[4] = {};

  stage(0, 0);
  wait_lds_copies();
  __syncthreads();

  int p = 0;
  for (int k0 = 0; k0 < K; k0 += 32) {
    if (k0 + 32 < K) stage(p ^ 1, k0 + 32);   // prefetch next tile (other buf)

    const __bf16 (*cA)[64][40]  = sA[p];
    const __bf16 (*cB)[128][40] = sB[p];

    // A fragments (16x32 bf16): lanes0-15 K=0..15, lanes16-31 K=16..31
    FragBF ah, al;
    {
      const int m  = wm0 + lr;
      const int kb = half * 16;
      ah.q[0] = *(const u32x4_t*)&cA[0][m][kb];
      ah.q[1] = *(const u32x4_t*)&cA[0][m][kb + 8];
      al.q[0] = *(const u32x4_t*)&cA[1][m][kb];
      al.q[1] = *(const u32x4_t*)&cA[1][m][kb + 8];
    }
    // B fragments (32x16 bf16): lanes0-15 K=0..15, lanes16-31 K=16..31
#pragma unroll
    for (int nc = 0; nc < 4; ++nc) {
      FragBF bh, bl;
      const int n  = wn0 + nc * 16 + lr;
      const int kb = half * 16;
      bh.q[0] = *(const u32x4_t*)&cB[0][n][kb];
      bh.q[1] = *(const u32x4_t*)&cB[0][n][kb + 8];
      bl.q[0] = *(const u32x4_t*)&cB[1][n][kb];
      bl.q[1] = *(const u32x4_t*)&cB[1][n][kb + 8];
      // compensated product: hi*hi + hi*lo + lo*hi  (lo*lo ~ 2^-34, dropped)
      acc[nc] = __builtin_amdgcn_wmma_f32_16x16x32_bf16(false, ah.v, false, bh.v,
                                                        (short)0, acc[nc], false, false);
      acc[nc] = __builtin_amdgcn_wmma_f32_16x16x32_bf16(false, ah.v, false, bl.v,
                                                        (short)0, acc[nc], false, false);
      acc[nc] = __builtin_amdgcn_wmma_f32_16x16x32_bf16(false, al.v, false, bh.v,
                                                        (short)0, acc[nc], false, false);
    }

    wait_lds_copies();   // own async copies for the NEXT tile are done
    __syncthreads();     // everyone finished reading buf p / writing buf p^1
    p ^= 1;
  }

  // ---- epilogue: C/D layout VGPR r -> M = r + 8*half, N = lr ----------------
#pragma unroll
  for (int nc = 0; nc < 4; ++nc) {
    const int col  = n0 + wn0 + nc * 16 + lr;
    const int row0 = m0 + wm0 + half * 8;
    if (STORE == 1) {
      // transposed split store: 8 consecutive rows per lane -> one 16B chunk
      Pack8 ph, pl;
#pragma unroll
      for (int r = 0; r < 8; ++r) {
        float v = acc[nc][r];
        if (BIAS) v += bias[(size_t)(row0 + r) * N + col];
        if (RELU) v = fmaxf(v, 0.0f);
        split2bf16(v, ph.h[r], pl.h[r]);
      }
      *(u32x4_t*)(outHi + (size_t)col * ldOutT + row0) = ph.q;
      *(u32x4_t*)(outLo + (size_t)col * ldOutT + row0) = pl.q;
    } else {
#pragma unroll
      for (int r = 0; r < 8; ++r) {
        const int row = row0 + r;
        if (row < M) {
          float v = acc[nc][r];
          if (BIAS) v += bias[(size_t)row * N + col];
          if (RELU) v = fmaxf(v, 0.0f);
          if (STORE == 0) outF[(size_t)row * N + col] = v;
          else            outF[(size_t)col * ldOutT + row] = v;  // transposed
        }
      }
    }
  }
}

// ===========================================================================
extern "C" void kernel_launch(void* const* d_in, const int* in_sizes, int n_in,
                              void* d_out, int out_size, void* d_ws, size_t ws_size,
                              hipStream_t stream) {
  (void)in_sizes; (void)n_in; (void)out_size; (void)ws_size;

  const float* U = (const float*)d_in[0];   // [4096, 2048]
  const float* A = (const float*)d_in[1];   // [1024, 1024]
  const float* B = (const float*)d_in[2];   // [1024, 2048]
  const float* C = (const float*)d_in[3];   // [1000, 1024]
  const float* D = (const float*)d_in[4];   // [1000, 2048]
  float* out = (float*)d_out;               // [4096, 1000]

  // ---- workspace layout (~121 MB) -----------------------------------------
  char* cur = (char*)d_ws;
  auto alloc = [&](size_t bytes) -> char* {
    char* p = cur;
    cur += (bytes + 255) & ~(size_t)255;
    return p;
  };
  __bf16* Aphi = (__bf16*)alloc(2ull * kHidden * kHidden);
  __bf16* Aplo = (__bf16*)alloc(2ull * kHidden * kHidden);
  __bf16* Bhi  = (__bf16*)alloc(2ull * kHidden * kInput);
  __bf16* Blo  = (__bf16*)alloc(2ull * kHidden * kInput);
  __bf16* Chi  = (__bf16*)alloc(2ull * kOutput * kHidden);
  __bf16* Clo  = (__bf16*)alloc(2ull * kOutput * kHidden);
  __bf16* Dhi  = (__bf16*)alloc(2ull * kOutput * kInput);
  __bf16* Dlo  = (__bf16*)alloc(2ull * kOutput * kInput);
  __bf16* Uhi  = (__bf16*)alloc(2ull * kBatch * kInput);
  __bf16* Ulo  = (__bf16*)alloc(2ull * kBatch * kInput);
  float*  BU   = (float*) alloc(4ull * kHidden * kBatch);   // [1024 x 4096]
  float*  T1   = (float*) alloc(4ull * kOutput * kBatch);   // [1000 x 4096]
  __bf16* XtHi[2]; __bf16* XtLo[2];                         // X^T: [4096 x 1024]
  for (int i = 0; i < 2; ++i) {
    XtHi[i] = (__bf16*)alloc(2ull * kBatch * kHidden);
    XtLo[i] = (__bf16*)alloc(2ull * kBatch * kHidden);
  }

  const dim3 blk(256);

  // 1) Ap = project_linf(A, 0.95), fused with bf16x2 split
  impdeq_project_split<<<kHidden, blk, 0, stream>>>(A, Aphi, Aplo, kHidden);
  // 2) bf16x2 splits of the static operands (all already K-contiguous)
  impdeq_split<<<2048, blk, 0, stream>>>(B, Bhi, Blo, (size_t)kHidden * kInput);
  impdeq_split<<<2048, blk, 0, stream>>>(C, Chi, Clo, (size_t)kOutput * kHidden);
  impdeq_split<<<2048, blk, 0, stream>>>(D, Dhi, Dlo, (size_t)kOutput * kInput);
  impdeq_split<<<4096, blk, 0, stream>>>(U, Uhi, Ulo, (size_t)kBatch * kInput);

  // 3) BU = B @ U^T   (M=1024, N=4096, K=2048)
  const dim3 gHid(kBatch / 128, kHidden / 64);
  impdeq_gemm<false, false, 0><<<gHid, blk, 0, stream>>>(
      Bhi, Blo, kInput, Uhi, Ulo, kInput, nullptr,
      BU, nullptr, nullptr, 0, kHidden, kBatch, kInput);

  // 4) X0^T = relu(BU)^T (first Picard step from X=0), split bf16x2
  impdeq_relu_split_T<<<4096, blk, 0, stream>>>(BU, XtHi[0], XtLo[0],
                                                kHidden, kBatch);

  // 5) Picard: X <- relu(Ap @ X + BU); X kept transposed & split.
  int buf = 0;
  for (int it = 0; it < kPicardIters; ++it) {
    const int nxt = buf ^ 1;
    impdeq_gemm<true, true, 1><<<gHid, blk, 0, stream>>>(
        Aphi, Aplo, kHidden, XtHi[buf], XtLo[buf], kHidden, BU,
        nullptr, XtHi[nxt], XtLo[nxt], kHidden, kHidden, kBatch, kHidden);
    buf = nxt;
  }

  // 6) T1 = D @ U^T   (M=1000, N=4096, K=2048)
  const dim3 gOut(kBatch / 128, (kOutput + 63) / 64);
  impdeq_gemm<false, false, 0><<<gOut, blk, 0, stream>>>(
      Dhi, Dlo, kInput, Uhi, Ulo, kInput, nullptr,
      T1, nullptr, nullptr, 0, kOutput, kBatch, kInput);

  // 7) out = (C @ X + T1)^T -> transposed f32 store into d_out [4096, 1000]
  impdeq_gemm<true, false, 2><<<gOut, blk, 0, stream>>>(
      Chi, Clo, kHidden, XtHi[buf], XtLo[buf], kHidden, T1,
      out, nullptr, nullptr, kOutput, kOutput, kBatch, kHidden);
}
